// RNN_38336878084708
// MI455X (gfx1250) — compile-verified, matrix operand order checked by reading, not run on hardware
//
#include <hip/hip_runtime.h>
#include <hip/hip_bf16.h>

// ---------------------------------------------------------------------------
// Problem constants (from reference)
// ---------------------------------------------------------------------------
#define BB   32
#define TT   35
#define KK   32
#define EE   64
#define DD   300
#define HH   4
#define CC   75
#define NHID 1024
#define HL   512
#define VG   40000
#define VS   25000
#define BT   (BB * TT)     // 1120
#define NEG_SLOPE 0.2f

typedef __attribute__((ext_vector_type(16))) __bf16 v16bf;
typedef __attribute__((ext_vector_type(4)))  __bf16 v4bf;
typedef __attribute__((ext_vector_type(8)))  float  v8f;
typedef __attribute__((ext_vector_type(4)))  float  v4f;

// ---------------------------------------------------------------------------
// WMMA helper: bf16 16x16x32, f32 accumulate
// ---------------------------------------------------------------------------
static __device__ inline v8f wmma_bf16(v16bf a, v16bf b, v8f c) {
    return __builtin_amdgcn_wmma_f32_16x16x32_bf16(
        false, a, false, b, (short)0, c, false, false);
}

// Fast unguarded fragment from row-major f32 (stride & k0 32-aligned):
// two pairs of float4 loads per lane -> global_load_b128 + v_cvt_pk_bf16_f32.
static __device__ inline v16bf load_frag_f32_fast(const float* base, int ld,
                                                  int row0, int k0, int lane) {
    const float* p = base + (size_t)(row0 + (lane & 15)) * ld + k0 + ((lane & 16) ? 8 : 0);
    v4f x0 = *(const v4f*)(p);
    v4f x1 = *(const v4f*)(p + 4);
    v4f x2 = *(const v4f*)(p + 16);
    v4f x3 = *(const v4f*)(p + 20);
    v16bf r;
#pragma unroll
    for (int j = 0; j < 4; ++j) {
        r[j]      = (__bf16)x0[j];
        r[j + 4]  = (__bf16)x1[j];
        r[j + 8]  = (__bf16)x2[j];
        r[j + 12] = (__bf16)x3[j];
    }
    return r;
}

// Fragment from row-major bf16 LDS: 8 contiguous bf16 per half -> ds_load_b128.
static __device__ inline v16bf load_frag_lds_bf(const __bf16* A, int lda, int row0,
                                                int k0, int lane) {
    v16bf r;
    const __bf16* p = A + (size_t)(row0 + (lane & 15)) * lda + k0 + ((lane & 16) ? 8 : 0);
#pragma unroll
    for (int i = 0; i < 8; ++i) { r[i] = p[i]; r[i + 8] = p[16 + i]; }
    return r;
}

static __device__ inline float sigm(float x) { return 1.0f / (1.0f + __expf(-x)); }

// ---------------------------------------------------------------------------
// GAT W-slab stager: Wst[64 n][32 k] = transpose of K-major gat_w slab.
// GUARD=false: pure coalesced loads; GUARD=true: clamp+select (branch-free).
// ---------------------------------------------------------------------------
template <bool GUARD>
static __device__ inline void gat_stage(const float* __restrict__ gat_w,
                                        __bf16* Wst, int tid, int n0, int k0) {
#pragma unroll
    for (int u4 = 0; u4 < 4; ++u4) {
        int u  = tid + 128 * u4;      // 0..511 quad units
        int n  = u & 63;
        int kq = (u >> 6) * 4;        // 0,4,...,28
        int gn = n0 + n;
        v4f v;
        if (!GUARD) {
#pragma unroll
            for (int j = 0; j < 4; ++j)
                v[j] = gat_w[(size_t)(k0 + kq + j) * DD + gn];
        } else {
            int ncl = (gn < DD) ? gn : (DD - 1);
#pragma unroll
            for (int j = 0; j < 4; ++j) {
                int k  = k0 + kq + j;
                int kc = (k < DD) ? k : (DD - 1);
                float x = gat_w[(size_t)kc * DD + ncl];
                v[j] = (k < DD && gn < DD) ? x : 0.f;
            }
        }
        v4bf bv = {(__bf16)v[0], (__bf16)v[1], (__bf16)v[2], (__bf16)v[3]};
        *(v4bf*)&Wst[n * 32 + kq] = bv;
    }
}

// ---------------------------------------------------------------------------
// Kernel 1: per-sample GAT -> sig[s,0:300]=word_emb, sig[s,300:600]=node_state
// Only attention segment dst==0 is needed (model keeps gat_out[:,0,:] only).
// ---------------------------------------------------------------------------
__global__ void gat_sig_kernel(const float* __restrict__ X,
                               const float* __restrict__ gat_w,
                               const float* __restrict__ a_src,
                               const float* __restrict__ a_dst,
                               const float* __restrict__ gat_bias,
                               const int*   __restrict__ node_idx,
                               const int*   __restrict__ edge_idx,
                               float*       __restrict__ sig) {
    __shared__ __bf16 xs[KK * 320];     // gathered features, K-padded to 320 (20 KB)
    __shared__ __bf16 xp[KK * DD];      // projected features 32x300       (19.2 KB)
    __shared__ __bf16 Wst[64 * 32];     // staged gat_w slab, N-MAJOR       (4 KB)
    __shared__ float  als[KK][HH], ald[KK][HH];
    __shared__ float  wE[EE + 1][HH];
    __shared__ int    sE[EE + 1];
    __shared__ unsigned char validE[EE + 4];

    const int s    = blockIdx.x;
    const int tid  = threadIdx.x;
    const int lane = tid & 31;
    const int wid  = tid >> 5;

    // ---- gather X rows -> bf16 LDS (row-strided, branch-free) ----
    for (int r = wid; r < KK; r += 4) {
        int nidx = node_idx[s * KK + r];
        const float* xr = X + (size_t)nidx * DD;
        for (int c = lane; c < DD; c += 32)       xs[r * 320 + c] = (__bf16)xr[c];
        for (int c = DD + lane; c < 320; c += 32) xs[r * 320 + c] = (__bf16)0.0f;
    }
    __syncthreads();

    // ---- xp = xs @ gat_w, 5 groups of 64 cols; W slab transposed into LDS ----
    for (int grp = 0; grp < 5; ++grp) {
        const int n0 = grp * 64;
        v8f acc0 = {}, acc1 = {};

        auto compute = [&](int k0) {
            __syncthreads();
            v16bf b  = load_frag_lds_bf(Wst, 32, wid * 16, 0, lane);
            v16bf a0 = load_frag_lds_bf(xs, 320, 0, k0, lane);
            v16bf a1 = load_frag_lds_bf(xs, 320, 16, k0, lane);
            acc0 = wmma_bf16(a0, b, acc0);
            acc1 = wmma_bf16(a1, b, acc1);
            __syncthreads();
        };

        if (n0 + 64 <= DD) {            // groups 0..3: branch-free main k-loop
            for (int k0 = 0; k0 + 32 <= DD; k0 += 32) {   // k0 = 0..256
                gat_stage<false>(gat_w, Wst, tid, n0, k0);
                compute(k0);
            }
        } else {                        // group 4: n-edge -> guarded staging
            for (int k0 = 0; k0 + 32 <= DD; k0 += 32) {
                gat_stage<true>(gat_w, Wst, tid, n0, k0);
                compute(k0);
            }
        }
        // tail k-slab (k = 288..319; zeros beyond 300)
        gat_stage<true>(gat_w, Wst, tid, n0, 288);
        compute(288);

        int n = n0 + wid * 16 + (lane & 15);
        if (n < DD) {
            int mb = (lane & 16) ? 8 : 0;
#pragma unroll
            for (int v = 0; v < 8; ++v) {
                xp[(mb + v) * DD + n]      = (__bf16)acc0[v];
                xp[(16 + mb + v) * DD + n] = (__bf16)acc1[v];
            }
        }
    }
    __syncthreads();

    // ---- per-node attention logits (128 threads = (k,h) pairs) ----
    {
        int k = tid >> 2, h = tid & 3;
        float as_ = 0.f, ad_ = 0.f;
        const __bf16* row = &xp[k * DD + h * CC];
        for (int c = 0; c < CC; ++c) {
            float v = (float)row[c];
            as_ += v * a_src[h * CC + c];
            ad_ += v * a_dst[h * CC + c];
        }
        als[k][h] = as_; ald[k][h] = ad_;
    }
    __syncthreads();

    // ---- edges into segment 0: E graph edges + self loop of node 0 ----
    if (tid < EE + 1) {
        int se, de;
        if (tid < EE) {
            se = edge_idx[(size_t)s * (2 * EE) + tid];
            de = edge_idx[(size_t)s * (2 * EE) + EE + tid];
        } else { se = 0; de = 0; }
        validE[tid] = (de == 0) ? 1 : 0;
        sE[tid] = se;
#pragma unroll
        for (int h = 0; h < HH; ++h) {
            float a = als[se][h] + ald[0][h];
            wE[tid][h] = (a > 0.f) ? a : NEG_SLOPE * a;   // leaky_relu
        }
    }
    __syncthreads();

    // ---- segment-0 softmax per head ----
    if (tid < HH) {
        int h = tid;
        float mx = -1e30f;
        for (int e = 0; e <= EE; ++e) if (validE[e]) mx = fmaxf(mx, wE[e][h]);
        float z = 0.f;
        for (int e = 0; e <= EE; ++e) if (validE[e]) z += __expf(wE[e][h] - mx);
        float inv = 1.f / (z + 1e-16f);
        for (int e = 0; e <= EE; ++e)
            wE[e][h] = validE[e] ? __expf(wE[e][h] - mx) * inv : 0.f;
    }
    __syncthreads();

    // ---- out row 0 (+bias) and word embedding -> sig ----
    int idx0 = node_idx[s * KK];
    for (int c = tid; c < DD; c += 128) {
        int h = c / CC;
        float acc = 0.f;
        for (int e = 0; e <= EE; ++e) acc += wE[e][h] * (float)xp[sE[e] * DD + c];
        sig[(size_t)s * (2 * DD) + DD + c] = acc + gat_bias[c];
        sig[(size_t)s * (2 * DD) + c]      = X[(size_t)idx0 * DD + c];
    }
}

// ---------------------------------------------------------------------------
// GEMM slab stager. KGUARD=false: pure b128 loads + prefetch (no clamps).
// n>=N rows are always zeroed via cndmask (cheap, needed only at the N edge).
// ---------------------------------------------------------------------------
template <bool KGUARD>
static __device__ inline void gemm_stage(const float* __restrict__ A,
                                         const float* __restrict__ W,
                                         __bf16* As, __bf16* Ws,
                                         int m0, int n0, int k0,
                                         int N, int K, int rA, int c4) {
    // ---- A slab (16 x 64): 2 quads per thread ----
#pragma unroll
    for (int half = 0; half < 2; ++half) {
        int kc4 = c4 + half * 32;
        v4f v;
        if (!KGUARD) {
            v = *(const v4f*)(A + (size_t)(m0 + rA) * K + k0 + kc4);
        } else {
#pragma unroll
            for (int q = 0; q < 4; ++q) {
                int k  = k0 + kc4 + q;
                int kk = (k < K) ? k : (K - 1);
                float x = A[(size_t)(m0 + rA) * K + kk];
                v[q] = (k < K) ? x : 0.f;
            }
        }
        v4bf bv = {(__bf16)v[0], (__bf16)v[1], (__bf16)v[2], (__bf16)v[3]};
        *(v4bf*)&As[rA * 64 + kc4] = bv;
    }
    // ---- W slab (64 x 64): 8 quads per thread; rows may exceed N ----
#pragma unroll
    for (int j = 0; j < 4; ++j) {
        int r  = rA + 16 * j;                    // 0..63
        int n  = n0 + r;
        int nr = (n < N) ? n : (N - 1);
#pragma unroll
        for (int half = 0; half < 2; ++half) {
            int kc4 = c4 + half * 32;
            const float* src = W + (size_t)nr * K + k0 + kc4;
            v4f v;
            if (!KGUARD) {
                v = *(const v4f*)src;
                __builtin_prefetch(src + 64, 0, 1);  // next k-slab
            } else {
#pragma unroll
                for (int q = 0; q < 4; ++q) {
                    int k  = k0 + kc4 + q;
                    int kk = (k < K) ? k : (K - 1);
                    float x = W[(size_t)nr * K + kk];
                    v[q] = (k < K) ? x : 0.f;
                }
            }
            if (n >= N) { v[0] = 0.f; v[1] = 0.f; v[2] = 0.f; v[3] = 0.f; }
            v4bf bv = {(__bf16)v[0], (__bf16)v[1], (__bf16)v[2], (__bf16)v[3]};
            *(v4bf*)&Ws[r * 64 + kc4] = bv;
        }
    }
}

// ---------------------------------------------------------------------------
// Kernel 2: C[MxN] = A[MxK] @ W[NxK]^T (+bias0+bias1). M % 16 == 0 assumed.
// 64-deep k-slabs staged in LDS (bf16): 2 WMMAs per barrier pair per wave.
// Branch-free main loop; single guarded tail block (only for K=600).
// ---------------------------------------------------------------------------
__global__ void gemm_nt_bias(const float* __restrict__ A,
                             const float* __restrict__ W,
                             const float* __restrict__ bias0,
                             const float* __restrict__ bias1,
                             float* __restrict__ C,
                             int M, int N, int K, int ldc) {
    __shared__ __bf16 As[16 * 64];      // 2 KB
    __shared__ __bf16 Ws[64 * 64];      // 8 KB
    const int tid  = threadIdx.x;
    const int lane = tid & 31;
    const int wid  = tid >> 5;
    const int m0   = blockIdx.y * 16;
    const int n0   = blockIdx.x * 64;

    const int rA = tid >> 3;            // 0..15
    const int c4 = (tid & 7) * 4;       // 0..28 (quad base within 32)

    v8f acc = {};
    auto compute = [&]() {
        __syncthreads();
#pragma unroll
        for (int ks = 0; ks < 2; ++ks) {
            v16bf a = load_frag_lds_bf(As, 64, 0, ks * 32, lane);
            v16bf b = load_frag_lds_bf(Ws, 64, wid * 16, ks * 32, lane);
            acc = wmma_bf16(a, b, acc);
        }
        __syncthreads();
    };

    const int Kf = K & ~63;             // multiple of 64
    for (int k0 = 0; k0 < Kf; k0 += 64) {
        gemm_stage<false>(A, W, As, Ws, m0, n0, k0, N, K, rA, c4);
        compute();
    }
    if (Kf < K) {                       // K=600 tail (k = 576..639)
        gemm_stage<true>(A, W, As, Ws, m0, n0, Kf, N, K, rA, c4);
        compute();
    }

    int n  = n0 + wid * 16 + (lane & 15);
    int mb = m0 + ((lane & 16) ? 8 : 0);
    if (n < N) {
        float bs = bias0 ? bias0[n] : 0.f;
        if (bias1) bs += bias1[n];
#pragma unroll
        for (int v = 0; v < 8; ++v)
            C[(size_t)(mb + v) * ldc + n] = acc[v] + bs;   // M % 16 == 0: no m guard
    }
}

// ---------------------------------------------------------------------------
// Kernel 3: one LSTM time step. hid % 32 == 0 -> fully unguarded b128 loads.
// Wave g computes gate g for a 16-col slice; fused sigmoid/tanh cell update.
// ---------------------------------------------------------------------------
__global__ void lstm_step(const float* __restrict__ xgate,
                          const float* __restrict__ whh,
                          const float* __restrict__ h_in,
                          float* __restrict__ h_out,
                          float* __restrict__ c_state,
                          float* __restrict__ ys,
                          int hid, int t, int sb, int st) {
    __shared__ float zbuf[4][BB][16];
    const int lane = threadIdx.x & 31;
    const int g    = threadIdx.x >> 5;          // gate: 0=i 1=f 2=g 3=o
    const int col0 = blockIdx.x * 16;
    const int ld4  = 4 * hid;

#pragma unroll
    for (int mt = 0; mt < 2; ++mt) {            // 32 batch rows = 2 M-tiles
        v8f acc = {};
        for (int k0 = 0; k0 < hid; k0 += 32) {
            v16bf a = load_frag_f32_fast(h_in, hid, mt * 16, k0, lane);
            v16bf b = load_frag_f32_fast(whh, hid, g * hid + col0, k0, lane);
            acc = wmma_bf16(a, b, acc);
        }
        int n  = lane & 15;
        int mb = mt * 16 + ((lane & 16) ? 8 : 0);
#pragma unroll
        for (int v = 0; v < 8; ++v) {
            int br = mb + v;
            float xg = xgate[(size_t)(br * TT + t) * ld4 + g * hid + col0 + n];
            zbuf[g][br][n] = acc[v] + xg;
        }
    }
    __syncthreads();

    for (int idx = threadIdx.x; idx < BB * 16; idx += 128) {
        int br = idx >> 4, n = idx & 15;
        float zi = zbuf[0][br][n], zf = zbuf[1][br][n];
        float zg = zbuf[2][br][n], zo = zbuf[3][br][n];
        int coff = br * hid + col0 + n;
        float cn = sigm(zf) * c_state[coff] + sigm(zi) * tanhf(zg);
        float h  = sigm(zo) * tanhf(cn);
        c_state[coff] = cn;
        h_out[coff]   = h;
        int row = br * sb + t * st;             // main: b*T+t ; senses final: t*B+b
        ys[(size_t)row * hid + col0 + n] = h;
    }
}

// ---------------------------------------------------------------------------
// Kernel 4: in-place row log-softmax
// ---------------------------------------------------------------------------
__global__ void log_softmax_kernel(float* __restrict__ logits, int V) {
    __shared__ float red[256];
    float* p = logits + (size_t)blockIdx.x * V;
    const int tid = threadIdx.x;

    float mx = -1e30f;
    for (int i = tid; i < V; i += 256) mx = fmaxf(mx, p[i]);
    red[tid] = mx; __syncthreads();
    for (int s2 = 128; s2 > 0; s2 >>= 1) {
        if (tid < s2) red[tid] = fmaxf(red[tid], red[tid + s2]);
        __syncthreads();
    }
    mx = red[0]; __syncthreads();

    float sum = 0.f;
    for (int i = tid; i < V; i += 256) sum += __expf(p[i] - mx);
    red[tid] = sum; __syncthreads();
    for (int s2 = 128; s2 > 0; s2 >>= 1) {
        if (tid < s2) red[tid] += red[tid + s2];
        __syncthreads();
    }
    float lse = logf(red[0]) + mx; __syncthreads();

    for (int i = tid; i < V; i += 256) p[i] -= lse;
}

// ---------------------------------------------------------------------------
// Orchestration
// ---------------------------------------------------------------------------
extern "C" void kernel_launch(void* const* d_in, const int* in_sizes, int n_in,
                              void* d_out, int out_size, void* d_ws, size_t ws_size,
                              hipStream_t stream) {
    (void)in_sizes; (void)n_in; (void)out_size; (void)ws_size;

    const float* X        = (const float*)d_in[0];
    const float* gat_w    = (const float*)d_in[1];
    const float* a_src    = (const float*)d_in[2];
    const float* a_dst    = (const float*)d_in[3];
    const float* gat_bias = (const float*)d_in[4];
    // main_lstm: 5..16 ; senses_lstm: 17..28 (wih, whh, bih, bhh per layer)
    const float* w_glob   = (const float*)d_in[29];
    const float* b_glob   = (const float*)d_in[30];
    const float* w_sens   = (const float*)d_in[31];
    const float* b_sens   = (const float*)d_in[32];
    const int*   node_idx = (const int*)d_in[33];
    const int*   edge_idx = (const int*)d_in[34];

    // workspace layout (f32)
    float* ws = (float*)d_ws;
    size_t o = 0;
    float* sig        = ws + o; o += (size_t)BT * (2 * DD);
    float* bufA       = ws + o; o += (size_t)BT * NHID;
    float* bufB       = ws + o; o += (size_t)BT * NHID;
    float* xgate      = ws + o; o += (size_t)BT * (4 * NHID);
    float* h_ping     = ws + o; o += (size_t)BB * NHID;
    float* h_pong     = ws + o; o += (size_t)BB * NHID;
    float* c_state    = ws + o; o += (size_t)BB * NHID;
    float* main_out   = ws + o; o += (size_t)BT * HL;
    float* senses_out = ws + o; o += (size_t)BT * HL;

    // ---- Phase 1: GAT + signal assembly ----
    gat_sig_kernel<<<BT, 128, 0, stream>>>(X, gat_w, a_src, a_dst, gat_bias,
                                           node_idx, edge_idx, sig);

    // ---- Phase 2: two 3-layer LSTM stacks ----
    const int hids[3] = {NHID, NHID, HL};
    const int inds[3] = {2 * DD, NHID, NHID};
    for (int stack = 0; stack < 2; ++stack) {
        int base = 5 + stack * 12;
        const float* act = sig;
        for (int l = 0; l < 3; ++l) {
            const int hid = hids[l], indim = inds[l], N4 = 4 * hid;
            const float* wih = (const float*)d_in[base + l * 4 + 0];
            const float* whh = (const float*)d_in[base + l * 4 + 1];
            const float* bih = (const float*)d_in[base + l * 4 + 2];
            const float* bhh = (const float*)d_in[base + l * 4 + 3];

            // xgate = act @ wih^T + bih + bhh  (all T steps at once)
            dim3 gg((N4 + 63) / 64, BT / 16);
            gemm_nt_bias<<<gg, 128, 0, stream>>>(act, wih, bih, bhh,
                                                 xgate, BT, N4, indim, N4);

            hipMemsetAsync(h_ping, 0, (size_t)BB * hid * sizeof(float), stream);
            hipMemsetAsync(c_state, 0, (size_t)BB * hid * sizeof(float), stream);

            float* ys; int sb = TT, st = 1;
            if (l == 2) {
                ys = stack ? senses_out : main_out;
                if (stack) { sb = 1; st = BB; }   // senses: row = t*B + b
            } else {
                ys = (l == 0) ? bufA : bufB;
            }

            for (int t = 0; t < TT; ++t) {
                float* hin  = (t & 1) ? h_pong : h_ping;
                float* hout = (t & 1) ? h_ping : h_pong;
                lstm_step<<<hid / 16, 128, 0, stream>>>(xgate, whh, hin, hout,
                                                        c_state, ys, hid, t, sb, st);
            }
            act = ys;
        }
    }

    // ---- Phase 3: vocab projections + log-softmax, straight into d_out ----
    float* out_glob = (float*)d_out;
    float* out_sens = out_glob + (size_t)BT * VG;

    gemm_nt_bias<<<dim3((VG + 63) / 64, BT / 16), 128, 0, stream>>>(
        main_out, w_glob, b_glob, nullptr, out_glob, BT, VG, HL, VG);
    log_softmax_kernel<<<BT, 256, 0, stream>>>(out_glob, VG);

    gemm_nt_bias<<<dim3((VS + 63) / 64, BT / 16), 128, 0, stream>>>(
        senses_out, w_sens, b_sens, nullptr, out_sens, BT, VS, HL, VS);
    log_softmax_kernel<<<BT, 256, 0, stream>>>(out_sens, VS);
}
